// MEM_CNN_RIEDEL_6116033429866
// MI455X (gfx1250) — compile-verified, matrix operand order checked by reading, not run on hardware
//
#include <hip/hip_runtime.h>
#include <hip/hip_bf16.h>
#include <math.h>

typedef __attribute__((ext_vector_type(16))) _Float16 v16h;
typedef __attribute__((ext_vector_type(8)))  _Float16 v8h;
typedef __attribute__((ext_vector_type(8)))  float    v8f;

// ---------------------------------------------------------------------------
// One 16x16x32 f16->f32 WMMA tile from LDS.
// A: row-major [m][k], lda halves.  Bn: column-major-as-rows [n][k], ldb halves.
// Per CDNA5 ISA 7.12.2 16-bit A layout: VGPR0-3 = K base..base+7,
// VGPR4-7 = K base+16..base+23, lanes 16-31 add +8. Both operands therefore
// load as two contiguous 16-byte chunks per lane (ds_load_b128).
// Caller guarantees 16B alignment: lda*2 % 16 == 0, base ptr 16B aligned.
// ---------------------------------------------------------------------------
__device__ __forceinline__ v8f wmma_tile(const _Float16* __restrict__ A, int lda,
                                         const _Float16* __restrict__ Bn, int ldb,
                                         v8f acc) {
  const int lane = threadIdx.x & 31;
  const int half = lane >> 4;
  const int mn   = lane & 15;
  const _Float16* ap = A + mn * lda + 8 * half;
  const _Float16* bp = Bn + mn * ldb + 8 * half;
  v8h a0 = *(const v8h*)ap;
  v8h a1 = *(const v8h*)(ap + 16);
  v8h b0 = *(const v8h*)bp;
  v8h b1 = *(const v8h*)(bp + 16);
  v16h a = __builtin_shufflevector(a0, a1, 0, 1, 2, 3, 4, 5, 6, 7,
                                   8, 9, 10, 11, 12, 13, 14, 15);
  v16h b = __builtin_shufflevector(b0, b1, 0, 1, 2, 3, 4, 5, 6, 7,
                                   8, 9, 10, 11, 12, 13, 14, 15);
  return __builtin_amdgcn_wmma_f32_16x16x32_f16(false, a, false, b, (short)0, acc,
                                                false, false);
}

// ---------------------------------------------------------------------------
// Kernel A: fused embedding gather + im2col conv (K=3, pad 2) + maxpool + tanh
// One block per sentence (B*S = 3840). K index = k*64 + f, padded to 192 (+8).
// Dynamic LDS: featLds[64][72] f16 | aLds[64][200] f16 | bCol[112][200] f16
// ---------------------------------------------------------------------------
constexpr int LDF = 72;    // featLds row stride (halves)
constexpr int LDA = 200;   // aLds / bCol row stride (halves); 400B = 25*16 OK
constexpr size_t OFF_FEAT = 0;
constexpr size_t OFF_AIM  = OFF_FEAT + (size_t)64 * LDF * 2;   // 9216
constexpr size_t OFF_BW   = OFF_AIM + (size_t)64 * LDA * 2;    // 34816
constexpr size_t SMEM_A   = OFF_BW + (size_t)112 * LDA * 2;    // 79616 B

__global__ __launch_bounds__(256) void sent_encoder(
    const int* __restrict__ tokens, const int* __restrict__ pos1,
    const int* __restrict__ pos2, const float* __restrict__ w2v,
    const float* __restrict__ pos_embed, const float* __restrict__ conv_w,
    const float* __restrict__ conv_b, const float* __restrict__ conv_bias,
    float* __restrict__ sent) {
  extern __shared__ char smemA[];
  _Float16* featLds = (_Float16*)(smemA + OFF_FEAT);  // [t][f], t=0..59 valid
  _Float16* aLds    = (_Float16*)(smemA + OFF_AIM);   // [t_out][K] im2col, zeros baked
  _Float16* bCol    = (_Float16*)(smemA + OFF_BW);    // [c][K]   weights, col-major B

  const int bs  = blockIdx.x;   // 0..3839
  const int tid = threadIdx.x;

  // gather features: word(50) + pos1(5) + pos2(5)
  for (int i = tid; i < 60 * 64; i += 256) {
    const int t = i >> 6, f = i & 63;
    float v = 0.f;
    if (f < 50)      v = w2v[(size_t)tokens[bs * 60 + t] * 50 + f];
    else if (f < 55) v = pos_embed[pos1[bs * 60 + t] * 5 + (f - 50)];
    else if (f < 60) v = pos_embed[pos2[bs * 60 + t] * 5 + (f - 55)];
    featLds[t * LDF + f] = (_Float16)v;
  }
  // weights as column-major B: bCol[c][K=k*64+f] = conv_w[c][f][k]
  for (int i = tid; i < 112 * LDA; i += 256) {
    const int n = i / LDA, kg = i % LDA;
    const int kk = kg >> 6, f = kg & 63;
    const float v = (kg < 192 && f < 60 && n < 100) ? conv_w[n * 180 + f * 3 + kk]
                                                    : 0.f;
    bCol[n * LDA + kg] = (_Float16)v;
  }
  __syncthreads();
  // im2col with boundary zeros baked in: aLds[t][k*64+f] = feat[t+k-2][f]
  for (int i = tid; i < 64 * LDA; i += 256) {
    const int t = i / LDA, kg = i % LDA;
    const int kk = kg >> 6, f = kg & 63;
    const int ts = t + kk - 2;
    const _Float16 v = (kg < 192 && f < 60 && ts >= 0 && ts < 60)
                           ? featLds[ts * LDF + f] : (_Float16)0.f;
    aLds[t * LDA + kg] = v;
  }
  __syncthreads();

  const int wave = tid >> 5;
  const int lane = tid & 31;
  if (wave < 7) {                       // wave w owns output channels 16w..16w+15
    const int nt   = wave;
    const int half = lane >> 4;
    const int mn   = lane & 15;
    float colmax = -3.0e38f;
    for (int mt = 0; mt < 4; ++mt) {    // output positions t = 16*mt .. 16*mt+15
      v8f acc = {};
#pragma unroll
      for (int kb = 0; kb < 6; ++kb)    // K = 192 in steps of 32
        acc = wmma_tile(aLds + (mt * 16) * LDA + kb * 32, LDA,
                        bCol + (nt * 16) * LDA + kb * 32, LDA, acc);
#pragma unroll
      for (int r = 0; r < 8; ++r) {            // C/D layout: M = r + 8*half
        const int t = mt * 16 + r + 8 * half;  // conv output position (< 62 valid)
        if (t < 62) colmax = fmaxf(colmax, acc[r]);
      }
    }
    colmax = fmaxf(colmax, __shfl_xor(colmax, 16, 32));  // merge M halves
    if (half == 0) {
      const int c = nt * 16 + mn;
      if (c < 100)
        sent[(size_t)bs * 100 + c] = tanhf(colmax + conv_b[c] + conv_bias[c]);
    }
  }
}

// ---------------------------------------------------------------------------
// Kernel B: qr[r,d] = sum_c r_embed[r,c] * att_W[r,c,d]   (R=53, C=100)
// ---------------------------------------------------------------------------
__global__ __launch_bounds__(128) void qr_kernel(const float* __restrict__ r_embed,
                                                 const float* __restrict__ att_W,
                                                 float* __restrict__ qr) {
  const int r = blockIdx.x;      // 53
  const int d = threadIdx.x;     // 128, use <100
  if (d < 100) {
    float s = 0.f;
    for (int c = 0; c < 100; ++c)
      s += r_embed[r * 100 + c] * att_W[((size_t)r * 100 + c) * 100 + d];
    qr[r * 100 + d] = s;
  }
}

// ---------------------------------------------------------------------------
// Kernel C: per-bag attention + 3 memory hops + scores, all LDS-resident.
// One block per bag (B=128). Hop GEMM (mk = memory @ C_ws[h]^T) on WMMA:
// A = memory f16 [64][136] row-major; B = C_ws[h] f16 [112][136] as [n][k].
// ---------------------------------------------------------------------------
constexpr int    LDM = 136;                // memH / bQ row stride (halves), 272B
constexpr int    SENTB_ELEMS = 30 * 104;   // f32  (aliased by probL [53][56])
constexpr int    QUERY_ELEMS = 56 * 104;   // f32
constexpr int    MEMH_ELEMS  = 64 * LDM;   // f16  memory, zero-padded
constexpr int    BQ_ELEMS    = 112 * LDM;  // f16  C_ws[h] as [q][c]
constexpr int    MK_ELEMS    = 56 * 104;   // f32  (aliased by attL [53][32] early)
constexpr int    RED_ELEMS   = 128;        // f32  scratch
constexpr size_t OFF_SENTB = 0;
constexpr size_t OFF_QUERY = OFF_SENTB + (size_t)SENTB_ELEMS * 4;  // 12480
constexpr size_t OFF_MEMH  = OFF_QUERY + (size_t)QUERY_ELEMS * 4;  // 35776
constexpr size_t OFF_BQ    = OFF_MEMH + (size_t)MEMH_ELEMS * 2;    // 53184
constexpr size_t OFF_MK    = OFF_BQ + (size_t)BQ_ELEMS * 2;        // 83648
constexpr size_t OFF_RED   = OFF_MK + (size_t)MK_ELEMS * 4;        // 106944
constexpr size_t SMEM_C    = OFF_RED + (size_t)RED_ELEMS * 4;      // 107456 B

__global__ __launch_bounds__(256) void bag_kernel(
    const float* __restrict__ sent, const float* __restrict__ qr,
    const int* __restrict__ en_pairs, const float* __restrict__ en_embed,
    const float* __restrict__ phi_q, const float* __restrict__ r_embed,
    const float* __restrict__ r_bias, const float* __restrict__ C_ws,
    const float* __restrict__ M_embed, float* __restrict__ out) {
  extern __shared__ char smemRaw[];
  float*    sentB = (float*)(smemRaw + OFF_SENTB);   // [30][104]
  float*    probL = (float*)(smemRaw + OFF_SENTB);   // [53][56]  (after sentB dead)
  float*    query = (float*)(smemRaw + OFF_QUERY);   // [53][104]
  _Float16* memH  = (_Float16*)(smemRaw + OFF_MEMH); // [64][136]
  _Float16* bQ    = (_Float16*)(smemRaw + OFF_BQ);   // [112][136]
  float*    mk    = (float*)(smemRaw + OFF_MK);      // [53][104]
  float*    attL  = (float*)(smemRaw + OFF_MK);      // [53][32]  (before mk live)
  float*    red   = (float*)(smemRaw + OFF_RED);     // [128]

  const int b   = blockIdx.x;
  const int tid = threadIdx.x;

  // load this bag's sentence vectors; zero the padded f16 memory tile
  for (int i = tid; i < 30 * 100; i += 256) {
    const int s = i / 100, c = i % 100;
    sentB[s * 104 + c] = sent[((size_t)b * 30 + s) * 100 + c];
  }
  for (int i = tid; i < MEMH_ELEMS; i += 256) memH[i] = (_Float16)0.f;
  __syncthreads();

  // relation-wise attention logits over sentences
  for (int i = tid; i < 53 * 30; i += 256) {
    const int r = i / 30, s = i % 30;
    float acc = 0.f;
    for (int c = 0; c < 100; ++c) acc += qr[r * 100 + c] * sentB[s * 104 + c];
    attL[r * 32 + s] = acc;
  }
  __syncthreads();
  if (tid < 53) {  // softmax over s per relation
    float m = -3.0e38f;
    for (int s = 0; s < 30; ++s) m = fmaxf(m, attL[tid * 32 + s]);
    float sum = 0.f;
    for (int s = 0; s < 30; ++s) {
      const float e = __expf(attL[tid * 32 + s] - m);
      attL[tid * 32 + s] = e; sum += e;
    }
    const float inv = 1.f / sum;
    for (int s = 0; s < 30; ++s) attL[tid * 32 + s] *= inv;
  }
  __syncthreads();
  // memory[r,d] = 0.5 * sum_s p * sent  (stored f16 for the hop WMMA)
  for (int i = tid; i < 53 * 100; i += 256) {
    const int r = i / 100, d = i % 100;
    float acc = 0.f;
    for (int s = 0; s < 30; ++s) acc += attL[r * 32 + s] * sentB[s * 104 + d];
    memH[r * LDM + d] = (_Float16)(0.5f * acc);
  }
  __syncthreads();

  // entity-pair query: qe[q] = (en0 + en1) @ phi_q ; query = r_embed + qe
  if (tid < 100) {
    const int e0 = en_pairs[b * 2 + 0], e1 = en_pairs[b * 2 + 1];
    float acc = 0.f;
    for (int c = 0; c < 50; ++c)
      acc += (en_embed[(size_t)e0 * 50 + c] + en_embed[(size_t)e1 * 50 + c]) *
             phi_q[c * 100 + tid];
    red[tid] = acc;
  }
  __syncthreads();
  for (int i = tid; i < 53 * 100; i += 256) {
    const int r = i / 100, q = i % 100;
    query[r * 104 + q] = r_embed[r * 100 + q] + red[q];
  }
  __syncthreads();

  // ---- memory hops ----
  const int wave = tid >> 5, lane = tid & 31;
  const int half = lane >> 4, mn = lane & 15;
  for (int h = 0; h < 3; ++h) {
    // B operand: C_ws[h] as [q][c] rows (already [n][k]); f16 with zero pad
    for (int i = tid; i < BQ_ELEMS; i += 256) {
      const int q = i / LDM, c = i % LDM;
      const float v = (q < 100 && c < 100) ? C_ws[h * 10000 + q * 100 + c] : 0.f;
      bQ[q * LDM + c] = (_Float16)v;
    }
    __syncthreads();
    // mk = memory @ C_ws^T  via WMMA: 4 M-tiles x 7 N-tiles, K=128 (4 steps)
    for (int tile = wave; tile < 28; tile += 8) {
      const int nt = tile / 4, mt = tile % 4;
      v8f acc = {};
#pragma unroll
      for (int kb = 0; kb < 4; ++kb)
        acc = wmma_tile(memH + (mt * 16) * LDM + kb * 32, LDM,
                        bQ + (nt * 16) * LDM + kb * 32, LDM, acc);
#pragma unroll
      for (int r = 0; r < 8; ++r) {
        const int m = mt * 16 + r + 8 * half;
        const int n = nt * 16 + mn;
        if (m < 53 && n < 100) mk[m * 104 + n] = acc[r];
      }
    }
    __syncthreads();
    // prob = softmax_s( query[r]·mk[s] ),  s over R=53
    for (int i = tid; i < 53 * 53; i += 256) {
      const int r = i / 53, s = i % 53;
      float acc = 0.f;
      for (int q = 0; q < 100; ++q) acc += query[r * 104 + q] * mk[s * 104 + q];
      probL[r * 56 + s] = acc;
    }
    __syncthreads();
    if (tid < 53) {
      float m = -3.0e38f;
      for (int s = 0; s < 53; ++s) m = fmaxf(m, probL[tid * 56 + s]);
      float sum = 0.f;
      for (int s = 0; s < 53; ++s) {
        const float e = __expf(probL[tid * 56 + s] - m);
        probL[tid * 56 + s] = e; sum += e;
      }
      const float inv = 1.f / sum;
      for (int s = 0; s < 53; ++s) probL[tid * 56 + s] *= inv;
    }
    __syncthreads();
    // query += prob @ mk
    for (int i = tid; i < 53 * 100; i += 256) {
      const int r = i / 100, q = i % 100;
      float acc = 0.f;
      for (int s = 0; s < 53; ++s) acc += probL[r * 56 + s] * mk[s * 104 + q];
      query[r * 104 + q] += acc;
    }
    __syncthreads();
  }

  // scores + log_softmax over relations
  if (tid < 53) {
    float acc = 0.f;
    for (int q = 0; q < 100; ++q) acc += query[tid * 104 + q] * M_embed[tid * 100 + q];
    red[tid] = acc + r_bias[tid];
  }
  __syncthreads();
  if (tid == 0) {
    float m = -3.0e38f;
    for (int r = 0; r < 53; ++r) m = fmaxf(m, red[r]);
    float sum = 0.f;
    for (int r = 0; r < 53; ++r) sum += __expf(red[r] - m);
    red[64] = m + __logf(sum);
  }
  __syncthreads();
  if (tid < 53) out[(size_t)b * 53 + tid] = red[tid] - red[64];
}

// ---------------------------------------------------------------------------
extern "C" void kernel_launch(void* const* d_in, const int* in_sizes, int n_in,
                              void* d_out, int out_size, void* d_ws, size_t ws_size,
                              hipStream_t stream) {
  (void)in_sizes; (void)n_in; (void)out_size; (void)ws_size;
  const int*   tokens    = (const int*)d_in[0];
  const int*   pos1      = (const int*)d_in[1];
  const int*   pos2      = (const int*)d_in[2];
  const int*   en_pairs  = (const int*)d_in[3];
  const float* w2v       = (const float*)d_in[4];
  const float* pos_embed = (const float*)d_in[5];
  const float* en_embed  = (const float*)d_in[6];
  const float* phi_q     = (const float*)d_in[7];
  const float* conv_w    = (const float*)d_in[8];
  const float* conv_b    = (const float*)d_in[9];
  const float* conv_bias = (const float*)d_in[10];
  const float* r_embed   = (const float*)d_in[11];
  const float* r_bias    = (const float*)d_in[12];
  const float* att_W     = (const float*)d_in[13];
  const float* C_ws      = (const float*)d_in[14];
  const float* M_embed   = (const float*)d_in[15];
  float* out  = (float*)d_out;
  float* sent = (float*)d_ws;              // 3840*100 f32 = 1.536 MB
  float* qrW  = sent + 3840 * 100;         // 53*100 f32

  sent_encoder<<<3840, 256, SMEM_A, stream>>>(tokens, pos1, pos2, w2v, pos_embed,
                                              conv_w, conv_b, conv_bias, sent);
  qr_kernel<<<53, 128, 0, stream>>>(r_embed, att_W, qrW);
  bag_kernel<<<128, 256, SMEM_C, stream>>>(sent, qrW, en_pairs, en_embed, phi_q,
                                           r_embed, r_bias, C_ws, M_embed, out);
}
